// MHA_85787676771080
// MI455X (gfx1250) — matrix operand order repaired and checked by reading, not compiled, not run on hardware
//
#include <hip/hip_runtime.h>
#include <hip/hip_bf16.h>

typedef __attribute__((ext_vector_type(16))) __bf16 v16bf;
typedef __attribute__((ext_vector_type(4)))  __bf16 v4bf;
typedef __attribute__((ext_vector_type(8)))  float  v8f;
typedef __attribute__((ext_vector_type(4)))  int    v4i;

union Frag {
    v16bf    v;
    unsigned u[8];
    __bf16   h[16];
};

#define DIM       1024
#define NQ        4096
#define NKV       4096
#define HEADS     16
#define HDIM      64
#define ATT_SCALE 0.125f   // 64^-0.5

static __device__ __forceinline__ v8f wmma_bf16(const v16bf& a, const v16bf& b, v8f c) {
    return __builtin_amdgcn_wmma_f32_16x16x32_bf16(false, a, false, b, (short)0, c, false, false);
}

// ---------------------------------------------------------------------------
// gfx1250 async global->LDS copy path (ASYNCcnt tracked), with safe fallback.
// Builtin signature (from clang diagnostic): (v4i global*, v4i local*, Ii, Ii)
// ---------------------------------------------------------------------------
#if __has_builtin(__builtin_amdgcn_global_load_async_to_lds_b128)
#define CDNA5_ASYNC 1
typedef __attribute__((address_space(1))) v4i gv4i_t;
typedef __attribute__((address_space(3))) v4i lv4i_t;
#endif

static __device__ __forceinline__ void cp16_to_lds(const __bf16* g, __bf16* l) {
#ifdef CDNA5_ASYNC
    __builtin_amdgcn_global_load_async_to_lds_b128((gv4i_t*)g, (lv4i_t*)l, 0, 0);
#else
    *(uint4*)l = *(const uint4*)g;
#endif
}

static __device__ __forceinline__ void cp_wait() {
#ifdef CDNA5_ASYNC
 #if __has_builtin(__builtin_amdgcn_s_wait_asynccnt)
    __builtin_amdgcn_s_wait_asynccnt(0);
 #else
    asm volatile("s_wait_asynccnt 0x0" ::: "memory");
 #endif
#endif
}

// ---------------------------------------------------------------------------
// f32 -> bf16 convert, 4 elements/thread
// ---------------------------------------------------------------------------
__global__ void cvt_f32_bf16_kernel(const float4* __restrict__ in,
                                    v4bf* __restrict__ out, int n4) {
    int i = blockIdx.x * blockDim.x + threadIdx.x;
    if (i < n4) {
        float4 f = in[i];
        v4bf b;
        b[0] = (__bf16)f.x; b[1] = (__bf16)f.y;
        b[2] = (__bf16)f.z; b[3] = (__bf16)f.w;
        out[i] = b;
    }
}

// ---------------------------------------------------------------------------
// GEMM: C[M,N] = A[M,K] @ B[N,K]^T + bias[N]   (torch Linear semantics)
// A,B bf16 row-major; bias f32; C bf16 or f32.
// 256 threads (8 waves), tile 128(M) x 128(N), K-step 32, double-buffered LDS
// streamed via async global->LDS. Wave grid 4(M) x 2(N); wave tile 32x64 =
// 2x4 WMMA tiles -> 8 v_wmma per wave per K-step.
// Requires M%128==0, N%128==0, K%32==0.
// ---------------------------------------------------------------------------
__global__ __launch_bounds__(256) void gemm_bf16_kernel(
    const __bf16* __restrict__ A, const __bf16* __restrict__ B,
    const float* __restrict__ bias, void* __restrict__ C,
    int M, int N, int K, int c_is_f32)
{
    __shared__ __bf16 As[2][128][32];   // 16 KB
    __shared__ __bf16 Bs[2][128][32];   // 16 KB

    const int tid   = threadIdx.x;
    const int lane  = tid & 31;
    const int wave  = tid >> 5;      // 0..7
    const int wm    = wave >> 1;     // 0..3  -> M offset 32*wm
    const int wn    = wave & 1;      // 0..1  -> N offset 64*wn
    const int l15   = lane & 15;
    const int lhalf = (lane >> 4) & 1;

    const int bm = blockIdx.x * 128;
    const int bn = blockIdx.y * 128;

    // staging coords: 2 threads/row, 16 bf16 (two b128 chunks) each
    const int sr = tid >> 1;
    const int sc = (tid & 1) * 16;

    auto stage = [&](int buf, int k0) {
        const __bf16* ga = A + (size_t)(bm + sr) * K + k0 + sc;
        const __bf16* gb = B + (size_t)(bn + sr) * K + k0 + sc;
        cp16_to_lds(ga,     &As[buf][sr][sc]);
        cp16_to_lds(ga + 8, &As[buf][sr][sc + 8]);
        cp16_to_lds(gb,     &Bs[buf][sr][sc]);
        cp16_to_lds(gb + 8, &Bs[buf][sr][sc + 8]);
    };

    v8f acc[2][4];
    #pragma unroll
    for (int mi = 0; mi < 2; ++mi)
        #pragma unroll
        for (int ni = 0; ni < 4; ++ni) acc[mi][ni] = 0;

    const int nk = K / 32;
    stage(0, 0);
    cp_wait();
    __syncthreads();

    for (int kt = 0; kt < nk; ++kt) {
        const int cur = kt & 1;
        if (kt + 1 < nk) stage(cur ^ 1, (kt + 1) * 32);

        // --- A fragments (row=lane&15, 16 contiguous K per half-lane)
        Frag af[2];
        #pragma unroll
        for (int mi = 0; mi < 2; ++mi) {
            const int row = wm * 32 + mi * 16 + l15;
            const int kb  = lhalf * 8;
            #pragma unroll
            for (int p = 0; p < 8; ++p) {
                int kk = kb * 2 + (p << 1);
                af[mi].u[p] = *(const unsigned*)(&As[cur][row][kk]);
            }
        }
        // --- B fragments (col=lane&15, 16 contiguous K per half-lane)
        Frag bf[4];
        #pragma unroll
        for (int ni = 0; ni < 4; ++ni) {
            const int row = wn * 64 + ni * 16 + l15;
            bf[ni].v = *(const v16bf*)(&Bs[cur][row][lhalf * 16]);
        }

        #pragma unroll
        for (int mi = 0; mi < 2; ++mi)
            #pragma unroll
            for (int ni = 0; ni < 4; ++ni)
                acc[mi][ni] = wmma_bf16(af[mi].v, bf[ni].v, acc[mi][ni]);

        cp_wait();          // next tile resident in LDS
        __syncthreads();    // all waves done reading cur before it is restaged
    }

    // --- epilogue: C layout VGPR j -> row j (+8 upper half-lane), col=lane&15
    #pragma unroll
    for (int mi = 0; mi < 2; ++mi) {
        #pragma unroll
        for (int ni = 0; ni < 4; ++ni) {
            const int col = bn + wn * 64 + ni * 16 + l15;
            const float bv = bias[col];
            #pragma unroll
            for (int j = 0; j < 8; ++j) {
                const int row = bm + wm * 32 + mi * 16 + j + lhalf * 8;
                const float v = acc[mi][ni][j] + bv;
                if (c_is_f32) ((float*)C)[(size_t)row * N + col] = v;
                else          ((__bf16*)C)[(size_t)row * N + col] = (__bf16)v;
            }
        }
    }
}

// ---------------------------------------------------------------------------
// Flash attention (non-causal). One workgroup = (head, 128-row q block).
// 8 waves; each wave owns one 16-row M tile and the full 64-col kv block.
// K tile staged [kv][c] via async copies; V tile staged transposed [c][kv]
// so the PV B-fragment is a contiguous 32-byte LDS read.
// ---------------------------------------------------------------------------
__global__ __launch_bounds__(256) void attn_kernel(
    const __bf16* __restrict__ Qp, const __bf16* __restrict__ Kp,
    const __bf16* __restrict__ Vp, __bf16* __restrict__ Out)
{
    __shared__ __bf16 Ks[64][64];        // 8 KB, [kv_row][c]
    __shared__ __bf16 Vt[64][64];        // 8 KB, [c][kv_row]  (transposed)
    __shared__ __bf16 Ps[8][16][64];     // 16 KB, per-wave P tile

    const int head  = blockIdx.y;
    const int qb    = blockIdx.x;
    const int tid   = threadIdx.x;
    const int lane  = tid & 31;
    const int wave  = tid >> 5;          // 0..7
    const int l15   = lane & 15;
    const int lhalf = (lane >> 4) & 1;
    const int hcol  = head * HDIM;

    // Q fragments for this wave's 16 rows: 2 K-steps over c=64
    Frag aq[2];
    {
        const int qrow = qb * 128 + wave * 16 + l15;
        const __bf16* qbase = Qp + (size_t)qrow * DIM + hcol;
        #pragma unroll
        for (int ks = 0; ks < 2; ++ks)
            #pragma unroll
            for (int p = 0; p < 8; ++p) {
                int kk = ks * 32 + lhalf * 16 + (p << 1);
                aq[ks].u[p] = *(const unsigned*)(qbase + kk);
            }
    }

    v8f o[4];
    o[0] = 0; o[1] = 0; o[2] = 0; o[3] = 0;
    float m_i[8], l_i[8];
    #pragma unroll
    for (int j = 0; j < 8; ++j) { m_i[j] = -1e30f; l_i[j] = 0.f; }

    for (int kv0 = 0; kv0 < NKV; kv0 += 64) {
        // --- stage K (async, direct) and V (transposed): 512 chunks of 8 bf16
        for (int t = tid; t < 512; t += 256) {
            const int r = t >> 3;          // kv row 0..63
            const int c = (t & 7) * 8;     // c col
            cp16_to_lds(Kp + (size_t)(kv0 + r) * DIM + hcol + c, &Ks[r][c]);
            uint4 vv = *(const uint4*)(Vp + (size_t)(kv0 + r) * DIM + hcol + c);
            const __bf16* ve = (const __bf16*)&vv;
            #pragma unroll
            for (int e = 0; e < 8; ++e) Vt[c + e][r] = ve[e];
        }
        cp_wait();
        __syncthreads();

        // --- S = Q @ K^T  (contraction over c=64, N = kv cols)
        v8f s[4];
        s[0] = 0; s[1] = 0; s[2] = 0; s[3] = 0;
        #pragma unroll
        for (int nt = 0; nt < 4; ++nt) {
            #pragma unroll
            for (int ks = 0; ks < 2; ++ks) {
                Frag bk;
                bk.v = *(const v16bf*)(&Ks[nt * 16 + l15][ks * 32 + lhalf * 16]);
                s[nt] = wmma_bf16(aq[ks].v, bk.v, s[nt]);
            }
        }
        #pragma unroll
        for (int nt = 0; nt < 4; ++nt) s[nt] *= ATT_SCALE;

        // --- online softmax (row j lives in element j, across each 16-lane half)
        float corr[8];
        #pragma unroll
        for (int j = 0; j < 8; ++j) {
            float t = fmaxf(fmaxf(s[0][j], s[1][j]), fmaxf(s[2][j], s[3][j]));
            #pragma unroll
            for (int off = 1; off < 16; off <<= 1)
                t = fmaxf(t, __shfl_xor(t, off, 32));
            const float mnew = fmaxf(m_i[j], t);
            corr[j] = __expf(m_i[j] - mnew);
            m_i[j]  = mnew;
        }
        float rs[8];
        #pragma unroll
        for (int j = 0; j < 8; ++j) rs[j] = 0.f;
        #pragma unroll
        for (int nt = 0; nt < 4; ++nt)
            #pragma unroll
            for (int j = 0; j < 8; ++j) {
                const float p = __expf(s[nt][j] - m_i[j]);
                s[nt][j] = p;
                rs[j] += p;
            }
        #pragma unroll
        for (int j = 0; j < 8; ++j) {
            #pragma unroll
            for (int off = 1; off < 16; off <<= 1)
                rs[j] += __shfl_xor(rs[j], off, 32);
            l_i[j] = l_i[j] * corr[j] + rs[j];
        }
        #pragma unroll
        for (int nt = 0; nt < 4; ++nt)
            #pragma unroll
            for (int j = 0; j < 8; ++j) o[nt][j] *= corr[j];

        // --- round-trip P through per-wave LDS to reach A-fragment layout
        #pragma unroll
        for (int nt = 0; nt < 4; ++nt)
            #pragma unroll
            for (int j = 0; j < 8; ++j)
                Ps[wave][j + lhalf * 8][nt * 16 + l15] = (__bf16)s[nt][j];
        // same-wave DS ops stay ordered; compiler inserts s_wait_dscnt

        Frag ap[2];
        #pragma unroll
        for (int ks = 0; ks < 2; ++ks)
            #pragma unroll
            for (int p = 0; p < 8; ++p) {
                int kk = ks * 32 + lhalf * 16 + (p << 1);
                ap[ks].u[p] = *(const unsigned*)(&Ps[wave][l15][kk]);
            }

        // --- O += P @ V  (contraction over kv=64, N = c cols; Vt -> contiguous B)
        #pragma unroll
        for (int nt = 0; nt < 4; ++nt) {
            #pragma unroll
            for (int ks = 0; ks < 2; ++ks) {
                Frag bv;
                bv.v = *(const v16bf*)(&Vt[nt * 16 + l15][ks * 32 + lhalf * 16]);
                o[nt] = wmma_bf16(ap[ks].v, bv.v, o[nt]);
            }
        }
        __syncthreads();   // before next K/V tile overwrite
    }

    // --- finalize: divide by row sum, store bf16 into (token, DIM) buffer
    #pragma unroll
    for (int nt = 0; nt < 4; ++nt) {
        #pragma unroll
        for (int j = 0; j < 8; ++j) {
            const int row = qb * 128 + wave * 16 + j + lhalf * 8;
            const int col = hcol + nt * 16 + l15;
            Out[(size_t)row * DIM + col] = (__bf16)(o[nt][j] / l_i[j]);
        }
    }
}

// ---------------------------------------------------------------------------
// Launch
// ---------------------------------------------------------------------------
extern "C" void kernel_launch(void* const* d_in, const int* in_sizes, int n_in,
                              void* d_out, int out_size, void* d_ws, size_t ws_size,
                              hipStream_t stream) {
    const float* q   = (const float*)d_in[0];
    const float* kv  = (const float*)d_in[1];
    const float* q_w = (const float*)d_in[2];
    const float* q_b = (const float*)d_in[3];
    const float* k_w = (const float*)d_in[4];
    const float* k_b = (const float*)d_in[5];
    const float* v_w = (const float*)d_in[6];
    const float* v_b = (const float*)d_in[7];
    const float* o_w = (const float*)d_in[8];
    const float* o_b = (const float*)d_in[9];

    const size_t ACT = (size_t)NQ * DIM;     // 4,194,304 elems
    const size_t WEI = (size_t)DIM * DIM;    // 1,048,576 elems

    char* ws = (char*)d_ws;
    size_t off = 0;
    __bf16* qb16 = (__bf16*)(ws + off); off += ACT * 2;
    __bf16* kv16 = (__bf16*)(ws + off); off += ACT * 2;
    __bf16* wq16 = (__bf16*)(ws + off); off += WEI * 2;
    __bf16* wk16 = (__bf16*)(ws + off); off += WEI * 2;
    __bf16* wv16 = (__bf16*)(ws + off); off += WEI * 2;
    __bf16* wo16 = (__bf16*)(ws + off); off += WEI * 2;
    __bf16* Qp   = (__bf16*)(ws + off); off += ACT * 2;
    __bf16* Kp   = (__bf16*)(ws + off); off += ACT * 2;
    __bf16* Vp   = (__bf16*)(ws + off); off += ACT * 2;
    __bf16* AO   = (__bf16*)(ws + off); off += ACT * 2;

    // 1) f32 -> bf16 converts (vectorized x4)
    {
        const int T = 256;
        const int na = (int)(ACT / 4), nw = (int)(WEI / 4);
        cvt_f32_bf16_kernel<<<(na + T - 1) / T, T, 0, stream>>>((const float4*)q,   (v4bf*)qb16, na);
        cvt_f32_bf16_kernel<<<(na + T - 1) / T, T, 0, stream>>>((const float4*)kv,  (v4bf*)kv16, na);
        cvt_f32_bf16_kernel<<<(nw + T - 1) / T, T, 0, stream>>>((const float4*)q_w, (v4bf*)wq16, nw);
        cvt_f32_bf16_kernel<<<(nw + T - 1) / T, T, 0, stream>>>((const float4*)k_w, (v4bf*)wk16, nw);
        cvt_f32_bf16_kernel<<<(nw + T - 1) / T, T, 0, stream>>>((const float4*)v_w, (v4bf*)wv16, nw);
        cvt_f32_bf16_kernel<<<(nw + T - 1) / T, T, 0, stream>>>((const float4*)o_w, (v4bf*)wo16, nw);
    }

    // 2) Q/K/V projections (bf16 out)
    dim3 ggrid(NQ / 128, DIM / 128);
    gemm_bf16_kernel<<<ggrid, 256, 0, stream>>>(qb16, wq16, q_b, Qp, NQ, DIM, DIM, 0);
    gemm_bf16_kernel<<<ggrid, 256, 0, stream>>>(kv16, wk16, k_b, Kp, NKV, DIM, DIM, 0);
    gemm_bf16_kernel<<<ggrid, 256, 0, stream>>>(kv16, wv16, v_b, Vp, NKV, DIM, DIM, 0);

    // 3) Flash attention
    dim3 agrid(NQ / 128, HEADS);
    attn_kernel<<<agrid, 256, 0, stream>>>(Qp, Kp, Vp, AO);

    // 4) Output projection (f32 out -> d_out)
    gemm_bf16_kernel<<<ggrid, 256, 0, stream>>>(AO, wo16, o_b, (float*)d_out, NQ, DIM, DIM, 1);
}
